// AttentionPairBias_49486613184627
// MI455X (gfx1250) — matrix operand order repaired and checked by reading, not compiled, not run on hardware
//
#include <hip/hip_runtime.h>
#include <hip/hip_bf16.h>

// ---------------------------------------------------------------------------
// AttentionPairBias for MI455X (gfx1250, wave32, WMMA)
// B=2, N=1024, D=768, E=128, H=16, HD=48 (padded to 64)
// ---------------------------------------------------------------------------

typedef __bf16 bf16_t;
typedef __attribute__((ext_vector_type(16))) __bf16       v16bf;
typedef __attribute__((ext_vector_type(8)))  float        v8f;
typedef __attribute__((ext_vector_type(4)))  unsigned int v4u;

#define WMMA_BF16(A, B, C) \
  __builtin_amdgcn_wmma_f32_16x16x32_bf16(false, (A), false, (B), (short)0, (C), false, false)

static constexpr int Nn  = 1024;
static constexpr int Dd  = 768;
static constexpr int Ee  = 128;
static constexpr int Hh  = 16;
static constexpr int HD  = 48;
static constexpr int HDP = 64;   // padded head dim (2 x K=32 WMMA steps)

union BF16x16 {
  v16bf v;
  v4u   q[2];
};

// ---------------------------------------------------------------------------
// Zero-init padded bf16 Q/K/V region (12 MB)
// ---------------------------------------------------------------------------
__global__ void zero_kernel(v4u* __restrict__ p, int n16) {
  int i = blockIdx.x * blockDim.x + threadIdx.x;
  if (i < n16) { v4u z = {0u, 0u, 0u, 0u}; p[i] = z; }
}

// ---------------------------------------------------------------------------
// Generic wave-per-tile WMMA GEMM:  Y(2048x768) = X(2048x768) @ W(768x768)^T
// mode 0: Q = (X Wq^T + bq) * rsqrt(48) -> bf16 [b,h,n,64]
// mode 1: K = X Wk^T                     -> bf16 [b,h,n,64]
// mode 2: V = X Wv^T                     -> bf16 [b,h,64,n]   (transposed)
// mode 3: G = sigmoid(X Wg^T)            -> f32  [b,n,768]
// mode 4: out = X Wo^T                   -> f32  [b,n,768]
// ---------------------------------------------------------------------------
__global__ void gemm_wmma(const float* __restrict__ X, const float* __restrict__ W,
                          const float* __restrict__ bq, void* __restrict__ out, int mode) {
  const int wid  = threadIdx.x >> 5;
  const int lane = threadIdx.x & 31;
  const int li   = lane & 15;
  const int hi   = lane >> 4;            // 0 or 1

  const int NT   = Dd / 16;              // 48 column tiles
  const int tile = blockIdx.x * (blockDim.x >> 5) + wid;
  const int m0   = (tile / NT) * 16;
  const int n0   = (tile % NT) * 16;

  const float* arow = X + (size_t)(m0 + li) * Dd;
  const float* brow = W + (size_t)(n0 + li) * Dd;

  v8f acc = {0.f, 0.f, 0.f, 0.f, 0.f, 0.f, 0.f, 0.f};

  for (int kb = 0; kb < Dd; kb += 32) {
    BF16x16 a, b;
    const float* ap = arow + kb + hi * 16;     // A: lane holds 16 consecutive k of row m
#pragma unroll
    for (int c = 0; c < 8; ++c) {
      a.v[c]     = (bf16_t)ap[c];
      a.v[8 + c] = (bf16_t)ap[8 + c];
    }
    const float* bp = brow + kb + hi * 16;     // B: lane holds 16 consecutive k of column n
#pragma unroll
    for (int c = 0; c < 16; ++c) b.v[c] = (bf16_t)bp[c];
    acc = WMMA_BF16(a.v, b.v, acc);
  }

  const int n = n0 + li;                       // output column (C layout: lane = column)
  if (mode == 0 || mode == 1) {
    const float sc   = (mode == 0) ? 0.14433756729740643f : 1.0f;  // 1/sqrt(48)
    const float bias = (mode == 0) ? bq[n] : 0.0f;
    const int h = n / HD, dd = n % HD;
    bf16_t* dst = (bf16_t*)out;
#pragma unroll
    for (int r = 0; r < 8; ++r) {
      const int m  = m0 + r + hi * 8;
      const int bb = m >> 10, ii = m & 1023;
      dst[(((size_t)(bb * Hh + h) * Nn + ii) << 6) + dd] = (bf16_t)((acc[r] + bias) * sc);
    }
  } else if (mode == 2) {
    const int h = n / HD, dd = n % HD;
    bf16_t* dst = (bf16_t*)out;
#pragma unroll
    for (int r = 0; r < 8; ++r) {
      const int m  = m0 + r + hi * 8;
      const int bb = m >> 10, ii = m & 1023;
      dst[((size_t)((bb * Hh + h) * HDP + dd) << 10) + ii] = (bf16_t)acc[r];
    }
  } else if (mode == 3) {
    float* dst = (float*)out;
#pragma unroll
    for (int r = 0; r < 8; ++r) {
      const int m = m0 + r + hi * 8;
      dst[(size_t)m * Dd + n] = 1.0f / (1.0f + __expf(-acc[r]));
    }
  } else {
    float* dst = (float*)out;
#pragma unroll
    for (int r = 0; r < 8; ++r) {
      const int m = m0 + r + hi * 8;
      dst[(size_t)m * Dd + n] = acc[r];
    }
  }
}

// ---------------------------------------------------------------------------
// Pair bias: bias[b,h,i,j] = (LayerNorm_E(edge[b,i,j,:]) @ Wz)[h] - 1e6*(1-mask[b,j])
// Block = 64 threads (2 waves) handles one (b,i) and a 32-wide j block.
// LN via lane-pair shfl reduction; E->H projection via 4x WMMA (H==16 == one tile).
// Stores bf16 to halve intermediate traffic (64 MB instead of 128 MB).
// ---------------------------------------------------------------------------
__global__ void pair_bias_kernel(const float* __restrict__ edge,
                                 const int* __restrict__ mask,
                                 const float* __restrict__ ln_g,
                                 const float* __restrict__ ln_b,
                                 const float* __restrict__ Wz,
                                 bf16_t* __restrict__ biasB) {
  __shared__ bf16_t zn[32][Ee];                // 8 KB normalized bf16

  const int tid  = threadIdx.x;
  const int lane = tid & 31;
  const int wid  = tid >> 5;
  const int li   = lane & 15;
  const int hi   = lane >> 4;

  const int blk = blockIdx.x;                  // B*N*(N/32) blocks
  const int jb  = (blk & 31) * 32;
  const int bi  = blk >> 5;                    // b*N + i
  const int bb  = bi >> 10;
  const int ii  = bi & 1023;

  // --- stats + normalize: thread t -> pair p = t/2, elements e0 = (t&1)*64
  {
    const int p  = tid >> 1;
    const int e0 = (tid & 1) * 64;
    const float* src = edge + (((size_t)bi * Nn) + (jb + p)) * Ee + e0;
    float s = 0.f, ss = 0.f;
    float vals[64];
#pragma unroll
    for (int c = 0; c < 64; ++c) { float x = src[c]; vals[c] = x; s += x; ss += x * x; }
    s  += __shfl_xor(s, 1, 32);
    ss += __shfl_xor(ss, 1, 32);
    const float mean = s * (1.0f / 128.0f);
    const float var  = ss * (1.0f / 128.0f) - mean * mean;
    const float rstd = rsqrtf(var + 1e-5f);
#pragma unroll
    for (int c = 0; c < 64; ++c) {
      const int e = e0 + c;
      zn[p][e] = (bf16_t)((vals[c] - mean) * rstd * ln_g[e] + ln_b[e]);
    }
  }
  __syncthreads();

  // --- WMMA: A = zn (16 pairs x 32e), B = Wz (32e x 16h), 4 K-chunks
  v8f acc = {0.f, 0.f, 0.f, 0.f, 0.f, 0.f, 0.f, 0.f};
  const int prow = wid * 16 + li;
#pragma unroll
  for (int c = 0; c < 4; ++c) {
    BF16x16 a, b;
    const bf16_t* zp = &zn[prow][c * 32 + hi * 16];
    a.q[0] = *(const v4u*)(zp);
    a.q[1] = *(const v4u*)(zp + 8);
#pragma unroll
    for (int e = 0; e < 16; ++e) b.v[e] = (bf16_t)Wz[(c * 32 + hi * 16 + e) * Hh + li];
    acc = WMMA_BF16(a.v, b.v, acc);
  }

  // --- store: lane column = head; rows = local pair index
  const int h = li;
#pragma unroll
  for (int r = 0; r < 8; ++r) {
    const int p = wid * 16 + r + hi * 8;
    const int j = jb + p;
    const float v = acc[r] + (1.0f - (float)mask[bb * Nn + j]) * (-1.0e6f);
    biasB[(((size_t)(bb * Hh + h) * Nn + ii) << 10) + j] = (bf16_t)v;
  }
}

// ---------------------------------------------------------------------------
// Flash attention: one wave per (b, h, 16-row i-tile).
// Computes S^T = K @ Q^T so the score C-layout is directly the A-fragment
// layout for P @ V after bf16 pack. Online softmax over j (1024) in 32-chunks.
// Epilogue applies 1/l and the sigmoid gate, writing g*o (f32).
// ---------------------------------------------------------------------------
__global__ void attn_kernel(const bf16_t* __restrict__ Qp,
                            const bf16_t* __restrict__ Kp,
                            const bf16_t* __restrict__ Vt,
                            const bf16_t* __restrict__ biasB,
                            const float* __restrict__ G,
                            float* __restrict__ Og) {
  const int wid  = threadIdx.x >> 5;
  const int lane = threadIdx.x & 31;
  const int li   = lane & 15;
  const int hi   = lane >> 4;

  const int task = blockIdx.x * (blockDim.x >> 5) + wid;   // 0..2047
  const int b    = task >> 10;
  const int rem  = task & 1023;
  const int h    = rem >> 6;
  const int i0   = (rem & 63) << 4;
  const int bh   = b * Hh + h;

  // Q^T B-fragments (lane = column i, consecutive d per lane)
  BF16x16 qf0, qf1;
  {
    const bf16_t* qrow = Qp + (((size_t)bh * Nn) + i0 + li) * HDP;
    qf0.q[0] = *(const v4u*)(qrow + hi * 8);
    qf0.q[1] = *(const v4u*)(qrow + 16 + hi * 8);
    qf1.q[0] = *(const v4u*)(qrow + 32 + hi * 8);
    qf1.q[1] = *(const v4u*)(qrow + 48 + hi * 8);
  }

  v8f o0 = {0.f,0.f,0.f,0.f,0.f,0.f,0.f,0.f};
  v8f o1 = o0, o2 = o0;
  const v8f zacc = o0;
  float mstate = -1.0e30f, lstate = 0.0f;

  const bf16_t* krows = Kp + (size_t)bh * Nn * HDP;
  const bf16_t* brow  = biasB + (((size_t)bh * Nn) + i0 + li) * Nn;
  const bf16_t* vbase = Vt + (size_t)bh * HDP * Nn;

  for (int jb = 0; jb < Nn; jb += 32) {
    // ---- scores: S^T chunks (j in [jb,jb+16) and [jb+16,jb+32))
    BF16x16 ka, kc;
    const bf16_t* k0 = krows + (size_t)(jb + li) * HDP;
    ka.q[0] = *(const v4u*)(k0 + hi * 8);
    ka.q[1] = *(const v4u*)(k0 + 16 + hi * 8);
    kc.q[0] = *(const v4u*)(k0 + 32 + hi * 8);
    kc.q[1] = *(const v4u*)(k0 + 48 + hi * 8);
    v8f s0 = WMMA_BF16(ka.v, qf0.v, zacc);
    s0     = WMMA_BF16(kc.v, qf1.v, s0);

    const bf16_t* k1 = k0 + 16 * HDP;
    ka.q[0] = *(const v4u*)(k1 + hi * 8);
    ka.q[1] = *(const v4u*)(k1 + 16 + hi * 8);
    kc.q[0] = *(const v4u*)(k1 + 32 + hi * 8);
    kc.q[1] = *(const v4u*)(k1 + 48 + hi * 8);
    v8f s1 = WMMA_BF16(ka.v, qf0.v, zacc);
    s1     = WMMA_BF16(kc.v, qf1.v, s1);

    // ---- add pair bias (8 consecutive bf16 per lane per chunk)
    BF16x16 b0, b1;
    b0.q[0] = *(const v4u*)(brow + jb + hi * 8);
    b1.q[0] = *(const v4u*)(brow + jb + 16 + hi * 8);
    float sc0[8], sc1[8];
#pragma unroll
    for (int r = 0; r < 8; ++r) {
      sc0[r] = s0[r] + (float)b0.v[r];
      sc1[r] = s1[r] + (float)b1.v[r];
    }

    // ---- online softmax update (row i = lane&15; combine j-halves via xor16)
    float mx = sc0[0];
#pragma unroll
    for (int r = 0; r < 8; ++r) { mx = fmaxf(mx, sc0[r]); mx = fmaxf(mx, sc1[r]); }
    mx = fmaxf(mx, __shfl_xor(mx, 16, 32));
    const float mnew = fmaxf(mstate, mx);
    const float corr = __expf(mstate - mnew);
    float ls = 0.f;
#pragma unroll
    for (int r = 0; r < 8; ++r) {
      sc0[r] = __expf(sc0[r] - mnew); ls += sc0[r];
      sc1[r] = __expf(sc1[r] - mnew); ls += sc1[r];
    }
    ls += __shfl_xor(ls, 16, 32);
    lstate = lstate * corr + ls;
    mstate = mnew;

    // ---- rescale O accumulators (row i = r + hi*8 -> broadcast from lane r+hi*8)
#pragma unroll
    for (int r = 0; r < 8; ++r) {
      const float c = __shfl(corr, r + hi * 8, 32);
      o0[r] *= c; o1[r] *= c; o2[r] *= c;
    }

    // ---- P fragment: S^T C-layout == A-layout of P after bf16 pack
    BF16x16 pa;
#pragma unroll
    for (int r = 0; r < 8; ++r) { pa.v[r] = (bf16_t)sc0[r]; pa.v[8 + r] = (bf16_t)sc1[r]; }

    // ---- O += P @ V  (V transposed: lane column d, contiguous j)
    BF16x16 vf;
    const size_t joff = (size_t)jb + hi * 8;
    vf.q[0] = *(const v4u*)(vbase + ((size_t)(0 * 16 + li) << 10) + joff);
    vf.q[1] = *(const v4u*)(vbase + ((size_t)(0 * 16 + li) << 10) + joff + 16);
    o0 = WMMA_BF16(pa.v, vf.v, o0);
    vf.q[0] = *(const v4u*)(vbase + ((size_t)(1 * 16 + li) << 10) + joff);
    vf.q[1] = *(const v4u*)(vbase + ((size_t)(1 * 16 + li) << 10) + joff + 16);
    o1 = WMMA_BF16(pa.v, vf.v, o1);
    vf.q[0] = *(const v4u*)(vbase + ((size_t)(2 * 16 + li) << 10) + joff);
    vf.q[1] = *(const v4u*)(vbase + ((size_t)(2 * 16 + li) << 10) + joff + 16);
    o2 = WMMA_BF16(pa.v, vf.v, o2);
  }

  // ---- epilogue: o/l, gate by sigmoid(G), write f32 [b,n,768]
  float il[8];
#pragma unroll
  for (int r = 0; r < 8; ++r) il[r] = 1.0f / __shfl(lstate, r + hi * 8, 32);
#pragma unroll
  for (int r = 0; r < 8; ++r) {
    const int i = i0 + r + hi * 8;
    const size_t rowo = ((size_t)b * Nn + i) * Dd + h * HD;
    Og[rowo + 0 * 16 + li] = o0[r] * il[r] * G[rowo + 0 * 16 + li];
    Og[rowo + 1 * 16 + li] = o1[r] * il[r] * G[rowo + 1 * 16 + li];
    Og[rowo + 2 * 16 + li] = o2[r] * il[r] * G[rowo + 2 * 16 + li];
  }
}

// ---------------------------------------------------------------------------
// Launch
// ---------------------------------------------------------------------------
extern "C" void kernel_launch(void* const* d_in, const int* in_sizes, int n_in,
                              void* d_out, int out_size, void* d_ws, size_t ws_size,
                              hipStream_t stream) {
  (void)in_sizes; (void)n_in; (void)out_size; (void)ws_size;

  const float* node_embed = (const float*)d_in[0];
  const float* edge_embed = (const float*)d_in[1];
  const int*   node_mask  = (const int*)d_in[2];
  const float* k_in       = (const float*)d_in[3];
  const float* Wq         = (const float*)d_in[4];
  const float* bq         = (const float*)d_in[5];
  const float* Wk         = (const float*)d_in[6];
  const float* Wv         = (const float*)d_in[7];
  const float* Wg         = (const float*)d_in[8];
  const float* ln_g       = (const float*)d_in[9];
  const float* ln_b       = (const float*)d_in[10];
  const float* Wz         = (const float*)d_in[11];
  const float* Wo         = (const float*)d_in[12];

  char* ws = (char*)d_ws;
  bf16_t* Qp    = (bf16_t*)(ws + ((size_t)0  << 20));  // 4 MB  bf16 [b,h,n,64]
  bf16_t* Kp    = (bf16_t*)(ws + ((size_t)4  << 20));  // 4 MB  bf16 [b,h,n,64]
  bf16_t* Vt    = (bf16_t*)(ws + ((size_t)8  << 20));  // 4 MB  bf16 [b,h,64,n]
  float*  Gbuf  = (float*) (ws + ((size_t)12 << 20));  // 6 MB  f32  [b,n,768]
  bf16_t* biasB = (bf16_t*)(ws + ((size_t)18 << 20));  // 64 MB bf16 [b,h,i,j]
  float*  Og    = (float*) (ws + ((size_t)82 << 20));  // 6 MB  f32  [b,n,768]

  // zero the padded bf16 Q/K/V region (12 MB = 786432 x 16B)
  zero_kernel<<<3072, 256, 0, stream>>>((v4u*)ws, 786432);

  // projections: 6144 16x16 tiles -> 768 blocks x 8 waves
  gemm_wmma<<<768, 256, 0, stream>>>(node_embed, Wq, bq, (void*)Qp, 0);
  gemm_wmma<<<768, 256, 0, stream>>>(k_in,       Wk, bq, (void*)Kp, 1);
  gemm_wmma<<<768, 256, 0, stream>>>(k_in,       Wv, bq, (void*)Vt, 2);
  gemm_wmma<<<768, 256, 0, stream>>>(node_embed, Wg, bq, (void*)Gbuf, 3);

  // pair bias: B*N*(N/32) = 65536 blocks of 64 threads; edge read exactly once
  pair_bias_kernel<<<65536, 64, 0, stream>>>(edge_embed, node_mask, ln_g, ln_b, Wz, biasB);

  // flash attention: 2048 wave-tasks -> 256 blocks x 8 waves
  attn_kernel<<<256, 256, 0, stream>>>(Qp, Kp, Vt, biasB, Gbuf, Og);

  // output projection: (g*o) @ Wo^T -> d_out (f32)
  gemm_wmma<<<768, 256, 0, stream>>>(Og, Wo, bq, d_out, 4);
}
